// MultiHeadAttention_89859305767740
// MI455X (gfx1250) — matrix operand order repaired and checked by reading, not compiled, not run on hardware
//
#include <hip/hip_runtime.h>
#include <hip/hip_bf16.h>

typedef __attribute__((ext_vector_type(16))) _Float16 v16h;
typedef __attribute__((ext_vector_type(8)))  _Float16 v8h;
typedef __attribute__((ext_vector_type(8)))  float    v8f;
typedef _Float16 half_t;

#define WMMA16x16x32(a, b, c) \
  __builtin_amdgcn_wmma_f32_16x16x32_f16(false, (a), false, (b), (short)0, (c), false, false)

// -------- optional CDNA5 async global->LDS path (guarded; falls back) -------
#if __has_builtin(__builtin_amdgcn_global_load_async_to_lds_b128)
#define HAS_ASYNC_LDS 1
#else
#define HAS_ASYNC_LDS 0
#endif

#if HAS_ASYNC_LDS
typedef int v4i_t __attribute__((vector_size(16)));
typedef __attribute__((address_space(1))) v4i_t* gas1_t;
typedef __attribute__((address_space(3))) v4i_t* las3_t;
__device__ __forceinline__ void async_cp16(void* lds, const void* g) {
  // 16B per lane, ASYNCcnt-tracked DMA into LDS
  __builtin_amdgcn_global_load_async_to_lds_b128(
      (gas1_t)(uintptr_t)g, (las3_t)(uint32_t)(uintptr_t)lds, 0, 0);
}
__device__ __forceinline__ void wait_async0() {
#if __has_builtin(__builtin_amdgcn_s_wait_asynccnt)
  __builtin_amdgcn_s_wait_asynccnt(0);
#else
  asm volatile("s_wait_asynccnt 0x0" ::: "memory");
#endif
}
#endif

namespace {
constexpr int Bn  = 4;
constexpr int Sn  = 2048;
constexpr int Dn  = 1024;
constexpr int Hn  = 16;
constexpr int HDn = 64;
constexpr int TOK = Bn * Sn;   // 8192

// GEMM tiling: block = 128x128, K-step 32, 8 waves (4m x 2n), wave = 32x64
constexpr int BM = 128, BN = 128, BK = 32;
constexpr int LDAs = 40;   // halves; BK + pad, multiple of 8 for 16B alignment
constexpr int LDBs = 40;

// attention tiling
constexpr int QB   = 128;  // queries per block (8 waves x 16)
constexpr int KBLK = 64;   // keys per LDS stage
constexpr int LKs  = 72;   // halves; HD + pad, multiple of 8

union V16 { v16h v; v8h h[2]; };
}  // namespace

// ---------------------------------------------------------------------------
// Double-buffered tiled GEMM body: C(128x128) += A(128xK) * W(KxN), f16 WMMA.
// A is f32 (converted on the fly) or f16 (async-copied to LDS when possible);
// W is f32, converted and transposed into LDS for contiguous B-fragments.
// One barrier per K-step; next tile's global traffic overlaps the 8 WMMAs.
// ---------------------------------------------------------------------------
template <bool A_HALF>
__device__ __forceinline__ void gemm_tile(const void* __restrict__ Ap,
                                          const float* __restrict__ W,
                                          v8f acc[2][4], int m0, int n0) {
  __shared__ __align__(16) half_t As[2][BM * LDAs];
  __shared__ __align__(16) half_t Bt[2][BN * LDBs];

  const int tid  = threadIdx.x;
  const int lane = tid & 31;
  const int wid  = tid >> 5;
  const int wm   = (wid & 3) * 32;
  const int wn   = (wid >> 2) * 64;
  const int lh   = lane & 15;
  const bool hi  = lane >= 16;

  for (int i = 0; i < 2; ++i)
    for (int j = 0; j < 4; ++j)
      for (int e = 0; e < 8; ++e) acc[i][j][e] = 0.0f;

  const int arow = tid >> 1;            // 0..127
  const int aseg = (tid & 1) * 16;      // 0 or 16
  const int bk   = lane;                // k within tile: 0..31
  const int bns  = wid * 16;            // n segment: 0..112

  union F16 { float4 q[4]; float f[16]; };
  F16 wreg, areg;
  v8h aregh0, aregh1;
  (void)aregh0; (void)aregh1;

  auto fetch_w = [&](int k0) {
    const float* Wp = W + (size_t)(k0 + bk) * Dn + n0 + bns;
    wreg.q[0] = ((const float4*)Wp)[0];
    wreg.q[1] = ((const float4*)Wp)[1];
    wreg.q[2] = ((const float4*)Wp)[2];
    wreg.q[3] = ((const float4*)Wp)[3];
  };
  auto fetch_a = [&](int k0) {
    if (A_HALF) {
#if !HAS_ASYNC_LDS
      const half_t* A = (const half_t*)Ap + (size_t)(m0 + arow) * Dn + k0 + aseg;
      aregh0 = *(const v8h*)A;
      aregh1 = *(const v8h*)(A + 8);
#else
      (void)k0;
#endif
    } else {
      const float* A = (const float*)Ap + (size_t)(m0 + arow) * Dn + k0 + aseg;
      areg.q[0] = ((const float4*)A)[0];
      areg.q[1] = ((const float4*)A)[1];
      areg.q[2] = ((const float4*)A)[2];
      areg.q[3] = ((const float4*)A)[3];
    }
  };
#if HAS_ASYNC_LDS
  auto async_a = [&](int k0, int b) {
    if (A_HALF) {
      const half_t* A = (const half_t*)Ap + (size_t)(m0 + arow) * Dn + k0 + aseg;
      async_cp16(&As[b][arow * LDAs + aseg],     A);
      async_cp16(&As[b][arow * LDAs + aseg + 8], A + 8);
    }
  };
#endif
  auto store_stage = [&](int b) {
    for (int j = 0; j < 16; ++j) Bt[b][(bns + j) * LDBs + bk] = (half_t)wreg.f[j];
    if (A_HALF) {
#if !HAS_ASYNC_LDS
      *(v8h*)&As[b][arow * LDAs + aseg]     = aregh0;
      *(v8h*)&As[b][arow * LDAs + aseg + 8] = aregh1;
#endif
    } else {
      v8h h0, h1;
      for (int j = 0; j < 8; ++j) { h0[j] = (_Float16)areg.f[j]; h1[j] = (_Float16)areg.f[8 + j]; }
      *(v8h*)&As[b][arow * LDAs + aseg]     = h0;
      *(v8h*)&As[b][arow * LDAs + aseg + 8] = h1;
    }
  };

  const int NIT = Dn / BK;  // 32

  // prologue: stage tile 0 into buffer 0
#if HAS_ASYNC_LDS
  async_a(0, 0);
#endif
  fetch_w(0);
  fetch_a(0);
  store_stage(0);
#if HAS_ASYNC_LDS
  if (A_HALF) wait_async0();
#endif
  __syncthreads();

  for (int it = 0; it < NIT; ++it) {
    const int cur = it & 1, nxt = cur ^ 1;
    const int k1 = (it + 1) * BK;
    if (it + 1 < NIT) {
#if HAS_ASYNC_LDS
      async_a(k1, nxt);   // buffer nxt free since barrier at end of it-1
#endif
      fetch_w(k1);
      fetch_a(k1);
    }

    // ---- compute from buffer `cur` ----
    // A 16x32: lanes 0-15 hold M=lh, K 0..7 & 16..23; lanes 16-31: K 8..15 & 24..31
    const int koa = hi ? 8 : 0;
    v16h af[2];
    for (int i = 0; i < 2; ++i) {
      const int row = wm + i * 16 + lh;
      V16 u;
      u.h[0] = *(const v8h*)&As[cur][row * LDAs + koa];
      u.h[1] = *(const v8h*)&As[cur][row * LDAs + koa + 16];
      af[i] = u.v;
    }
    // B 32x16: lanes 0-15 hold N=lh, K 0..7 & 16..23; lanes 16-31: K 8..15 & 24..31
    const int kob = hi ? 8 : 0;
    v16h bf[4];
    for (int j = 0; j < 4; ++j) {
      const int col = wn + j * 16 + lh;
      V16 u;
      u.h[0] = *(const v8h*)&Bt[cur][col * LDBs + kob];
      u.h[1] = *(const v8h*)&Bt[cur][col * LDBs + kob + 16];
      bf[j] = u.v;
    }
    for (int i = 0; i < 2; ++i)
      for (int j = 0; j < 4; ++j)
        acc[i][j] = WMMA16x16x32(af[i], bf[j], acc[i][j]);

    if (it + 1 < NIT) {
      store_stage(nxt);
#if HAS_ASYNC_LDS
      if (A_HALF) wait_async0();
#endif
      __syncthreads();
    }
  }
}

// ---------------------------------------------------------------------------
// Kernel 1: Q/K/V projections (grid.z selects which), output f16 [B][H][S][64]
// ---------------------------------------------------------------------------
__global__ __launch_bounds__(256) void mha_qkv_proj(
    const float* __restrict__ xq, const float* __restrict__ xk, const float* __restrict__ xv,
    const float* __restrict__ Wq, const float* __restrict__ Wk, const float* __restrict__ Wv,
    const float* __restrict__ bq, const float* __restrict__ bk, const float* __restrict__ bv,
    half_t* __restrict__ qo, half_t* __restrict__ ko, half_t* __restrict__ vo) {
  const int which = blockIdx.z;
  const float* X  = (which == 0) ? xq : (which == 1) ? xk : xv;
  const float* W  = (which == 0) ? Wq : (which == 1) ? Wk : Wv;
  const float* bi = (which == 0) ? bq : (which == 1) ? bk : bv;
  half_t* Out     = (which == 0) ? qo : (which == 1) ? ko : vo;

  const int m0 = blockIdx.y * BM;
  const int n0 = blockIdx.x * BN;
  v8f acc[2][4];
  gemm_tile<false>(X, W, acc, m0, n0);

  const int lane = threadIdx.x & 31;
  const int wid  = threadIdx.x >> 5;
  const int wm = (wid & 3) * 32, wn = (wid >> 2) * 64;
  const int lh = lane & 15;
  const int radd = (lane < 16) ? 0 : 8;
  for (int j = 0; j < 4; ++j) {
    const int n = n0 + wn + j * 16 + lh;
    const int h = n >> 6, d = n & 63;
    const float bval = bi[n];
    for (int i = 0; i < 2; ++i)
      for (int r = 0; r < 8; ++r) {
        const int m = m0 + wm + i * 16 + r + radd;
        const int b = m >> 11, s = m & 2047;
        Out[(((size_t)(b * Hn + h)) * Sn + s) * HDn + d] = (half_t)(acc[i][j][r] + bval);
      }
  }
}

// ---------------------------------------------------------------------------
// Kernel 2: flash attention, double-buffered K/V staging. Scores computed
// transposed (St = K * Q^T) so softmaxed P lands directly in A-fragment layout.
// ---------------------------------------------------------------------------
__global__ __launch_bounds__(256) void mha_flash_attn(
    const half_t* __restrict__ Qw, const half_t* __restrict__ Kw,
    const half_t* __restrict__ Vw, const int* __restrict__ mask,
    half_t* __restrict__ attn_out) {
  __shared__ __align__(16) half_t Ks[2][KBLK * LKs];   // [key][d]
  __shared__ __align__(16) half_t Vt[2][HDn * LKs];    // [d][key] (transposed)
  __shared__ float madd[2][KBLK];

  const int bh = blockIdx.y;
  const int b  = bh >> 4;
  const int h  = bh & 15;
  const int q0 = blockIdx.x * QB;

  const half_t* Qb = Qw + (size_t)bh * Sn * HDn;
  const half_t* Kb = Kw + (size_t)bh * Sn * HDn;
  const half_t* Vb = Vw + (size_t)bh * Sn * HDn;
  const int* mb = mask + (size_t)b * Sn;

  const int tid  = threadIdx.x;
  const int lane = tid & 31, wid = tid >> 5;
  const int lh   = lane & 15;
  const bool hi  = lane >= 16;
  const int qw   = q0 + wid * 16;   // this wave's 16 queries

  // Q as B-operand (Q^T, 32d x 16q): lane q holds its row's d-values.
  v16h qf[2];
  for (int p = 0; p < 2; ++p) {
    const half_t* src = Qb + (size_t)(qw + lh) * HDn + p * 32 + (hi ? 8 : 0);
    V16 u;
    u.h[0] = *(const v8h*)src;
    u.h[1] = *(const v8h*)(src + 16);
    qf[p] = u.v;
  }

  v8f oacc[4];
  for (int t = 0; t < 4; ++t)
    for (int e = 0; e < 8; ++e) oacc[t][e] = 0.0f;
  float m_run = -1.0e30f, l_run = 0.0f;

  const int skey = tid & 63;
  const int sd0  = (tid >> 6) * 16;
  const int NB   = Sn / KBLK;   // 32

  // ---- prologue: stage block 0 into buffer 0 ----
  {
    const half_t* ksrc = Kb + (size_t)skey * HDn + sd0;
#if HAS_ASYNC_LDS
    async_cp16(&Ks[0][skey * LKs + sd0],     ksrc);
    async_cp16(&Ks[0][skey * LKs + sd0 + 8], ksrc + 8);
#else
    *(v8h*)&Ks[0][skey * LKs + sd0]     = *(const v8h*)ksrc;
    *(v8h*)&Ks[0][skey * LKs + sd0 + 8] = *(const v8h*)(ksrc + 8);
#endif
    const half_t* vsrc = Vb + (size_t)skey * HDn + sd0;
    v8h a = *(const v8h*)vsrc;
    v8h c = *(const v8h*)(vsrc + 8);
    for (int j = 0; j < 8; ++j) Vt[0][(sd0 + j) * LKs + skey]     = a[j];
    for (int j = 0; j < 8; ++j) Vt[0][(sd0 + 8 + j) * LKs + skey] = c[j];
    if (tid < KBLK) madd[0][tid] = (mb[tid] == 0) ? -1.0e9f : 0.0f;
#if HAS_ASYNC_LDS
    wait_async0();
#endif
  }
  __syncthreads();

  v8h ka, kc, va, vc;
  (void)ka; (void)kc;
  int mval = 0;

  for (int it = 0; it < NB; ++it) {
    const int cur = it & 1, nxt = cur ^ 1;
    const int kb1 = (it + 1) * KBLK;
    if (it + 1 < NB) {
      const half_t* ksrc = Kb + (size_t)(kb1 + skey) * HDn + sd0;
#if HAS_ASYNC_LDS
      async_cp16(&Ks[nxt][skey * LKs + sd0],     ksrc);
      async_cp16(&Ks[nxt][skey * LKs + sd0 + 8], ksrc + 8);
#else
      ka = *(const v8h*)ksrc;
      kc = *(const v8h*)(ksrc + 8);
#endif
      const half_t* vsrc = Vb + (size_t)(kb1 + skey) * HDn + sd0;
      va = *(const v8h*)vsrc;
      vc = *(const v8h*)(vsrc + 8);
      mval = mb[kb1 + (tid & 63)];
    }

    // ---- scores (transposed): 4 tiles of [16 keys][16 queries] ----
    float P[4][8];
    float mblk = -1.0e30f;
    for (int st = 0; st < 4; ++st) {
      v8f sacc;
      for (int e = 0; e < 8; ++e) sacc[e] = 0.0f;
      for (int p = 0; p < 2; ++p) {
        const int key  = st * 16 + lh;
        const int doff = p * 32 + (hi ? 8 : 0);
        V16 u;
        u.h[0] = *(const v8h*)&Ks[cur][key * LKs + doff];
        u.h[1] = *(const v8h*)&Ks[cur][key * LKs + doff + 16];
        sacc = WMMA16x16x32(u.v, qf[p], sacc);
      }
      // C-layout: VGPR r = key st*16+r(+8 for hi lanes), lane = query
      const float* mp = &madd[cur][st * 16 + (hi ? 8 : 0)];
      for (int r = 0; r < 8; ++r) {
        const float s = sacc[r] * 0.125f + mp[r];   // 1/sqrt(64) then mask
        P[st][r] = s;
        mblk = fmaxf(mblk, s);
      }
    }
    // per-query online softmax (query lives in lane; combine both key halves)
    mblk = fmaxf(mblk, __shfl_xor(mblk, 16, 32));
    const float m_new = fmaxf(m_run, mblk);
    const float corr  = __expf(m_run - m_new);
    float rsum = 0.0f;
    for (int st = 0; st < 4; ++st)
      for (int r = 0; r < 8; ++r) {
        const float e = __expf(P[st][r] - m_new);
        P[st][r] = e;
        rsum += e;
      }
    rsum += __shfl_xor(rsum, 16, 32);
    l_run = l_run * corr + rsum;
    m_run = m_new;
    // rescale O (O C-layout: VGPR r = query row) via lane redistribution
    for (int r = 0; r < 8; ++r) {
      const float cr = __shfl(corr, hi ? (r + 8) : r, 32);
      for (int t = 0; t < 4; ++t) oacc[t][r] *= cr;
    }
    // P is already in A-fragment layout: pack f32 -> f16 and run P*V
    for (int p = 0; p < 2; ++p) {
      v16h pf;
      for (int j = 0; j < 8; ++j) {
        pf[j]     = (_Float16)P[2 * p][j];
        pf[8 + j] = (_Float16)P[2 * p + 1][j];
      }
      for (int t = 0; t < 4; ++t) {
        const int row  = t * 16 + lh;                // d-column of V
        const int koff = p * 32 + (hi ? 8 : 0);      // key offset
        V16 u;
        u.h[0] = *(const v8h*)&Vt[cur][row * LKs + koff];
        u.h[1] = *(const v8h*)&Vt[cur][row * LKs + koff + 16];
        oacc[t] = WMMA16x16x32(pf, u.v, oacc[t]);
      }
    }

    if (it + 1 < NB) {
#if !HAS_ASYNC_LDS
      *(v8h*)&Ks[nxt][skey * LKs + sd0]     = ka;
      *(v8h*)&Ks[nxt][skey * LKs + sd0 + 8] = kc;
#endif
      for (int j = 0; j < 8; ++j) Vt[nxt][(sd0 + j) * LKs + skey]     = va[j];
      for (int j = 0; j < 8; ++j) Vt[nxt][(sd0 + 8 + j) * LKs + skey] = vc[j];
      if (tid < KBLK) madd[nxt][tid] = (mval == 0) ? -1.0e9f : 0.0f;
#if HAS_ASYNC_LDS
      wait_async0();
#endif
      __syncthreads();
    }
  }

  const float rinv = 1.0f / l_run;
  for (int r = 0; r < 8; ++r) {
    const float sr = __shfl(rinv, hi ? (r + 8) : r, 32);
    const int q = qw + r + (hi ? 8 : 0);
    for (int t = 0; t < 4; ++t) {
      const int d = t * 16 + lh;
      attn_out[((size_t)(b * Sn + q)) * Dn + h * HDn + d] = (half_t)(oacc[t][r] * sr);
    }
  }
}

// ---------------------------------------------------------------------------
// Kernel 3: output projection, f16 attn -> f32 out
// ---------------------------------------------------------------------------
__global__ __launch_bounds__(256) void mha_out_proj(
    const half_t* __restrict__ attn, const float* __restrict__ Wo,
    const float* __restrict__ bo, float* __restrict__ out) {
  const int m0 = blockIdx.y * BM, n0 = blockIdx.x * BN;
  v8f acc[2][4];
  gemm_tile<true>(attn, Wo, acc, m0, n0);

  const int lane = threadIdx.x & 31, wid = threadIdx.x >> 5;
  const int wm = (wid & 3) * 32, wn = (wid >> 2) * 64;
  const int lh = lane & 15;
  const int radd = (lane < 16) ? 0 : 8;
  for (int j = 0; j < 4; ++j) {
    const int n = n0 + wn + j * 16 + lh;
    const float bval = bo[n];
    for (int i = 0; i < 2; ++i)
      for (int r = 0; r < 8; ++r) {
        const int m = m0 + wm + i * 16 + r + radd;
        out[(size_t)m * Dn + n] = acc[i][j][r] + bval;
      }
  }
}

// ---------------------------------------------------------------------------
extern "C" void kernel_launch(void* const* d_in, const int* in_sizes, int n_in,
                              void* d_out, int out_size, void* d_ws, size_t ws_size,
                              hipStream_t stream) {
  const float* query = (const float*)d_in[0];
  const float* key_  = (const float*)d_in[1];
  const float* value = (const float*)d_in[2];
  const int*   mask  = (const int*)d_in[3];
  const float* Wq = (const float*)d_in[4];
  const float* bq = (const float*)d_in[5];
  const float* Wk = (const float*)d_in[6];
  const float* bk = (const float*)d_in[7];
  const float* Wv = (const float*)d_in[8];
  const float* bv = (const float*)d_in[9];
  const float* Wo = (const float*)d_in[10];
  const float* bo = (const float*)d_in[11];

  half_t* ws = (half_t*)d_ws;
  const size_t QSZ = (size_t)Bn * Hn * Sn * HDn;   // 8.39M halves each
  half_t* qws = ws;
  half_t* kws = ws + QSZ;
  half_t* vws = ws + 2 * QSZ;
  half_t* aws = ws + 3 * QSZ;

  dim3 blk(256, 1, 1);
  dim3 g1(Dn / BN, TOK / BM, 3);
  mha_qkv_proj<<<g1, blk, 0, stream>>>(query, key_, value, Wq, Wk, Wv,
                                       bq, bk, bv, qws, kws, vws);
  dim3 g2(Sn / QB, Bn * Hn, 1);
  mha_flash_attn<<<g2, blk, 0, stream>>>(qws, kws, vws, mask, aws);
  dim3 g3(Dn / BN, TOK / BM, 1);
  mha_out_proj<<<g3, blk, 0, stream>>>(aws, Wo, bo, (float*)d_out);
}
